// tffts_unet_53214644797611
// MI455X (gfx1250) — matrix operand order repaired and checked by reading, not compiled, not run on hardware
//
#include <hip/hip_runtime.h>

// ---------------------------------------------------------------------------
// STFT (Hann window + 512-pt forward DFT) as split-f16 WMMA GEMM for gfx1250.
//   rx = F @ Cr,  ix = F @ Ci,   F: (32768 x 512) windowed frames
//   Cr[n][k] = cos(2*pi*n*k/512), Ci[n][k] = -sin(2*pi*n*k/512)
// Operands split x = hi + lo (f16); keep hi*hi + hi*lo + lo*hi (err ~1e-6).
// Round 1 -> 2: repartitioned waves (2 row tiles x 8 col tiles each) to cut
// live VGPRs ~in half and eliminate the scratch spills seen in the disasm.
// ---------------------------------------------------------------------------

typedef _Float16 v16h __attribute__((ext_vector_type(16)));
typedef _Float16 v8h  __attribute__((ext_vector_type(8)));
typedef float    v8f  __attribute__((ext_vector_type(8)));

#define NFFT   512
#define HLEN   128
#define BATCH  16
#define SLEN   262144
#define NUMF   2048
#define ROWS   (BATCH * NUMF)      // 32768 frames total
#define RPB    64                  // rows (frames) per block
#define LDSS   520                 // LDS row stride in halves (512 + 8 pad)
#define BF_MAT 262144              // halves per B-fragment matrix (16*32*32*16)

// ---------------------------------------------------------------------------
// Build Cr/Ci hi/lo directly in WMMA B-fragment order:
//   half index = ((ks*32 + ct)*32 + lane)*16 + i
//   K   = ks*32 + ((lane>>4)<<3) + (i&7) + ((i>>3)<<4)  (B layout: lanes 0-15 -> K 0..7,16..23,
//   col = ct*16 + (lane & 15)                lanes 16-31 -> K 8..15,24..31)
// ---------------------------------------------------------------------------
__global__ void build_bfrags(_Float16* __restrict__ ws) {
    int id   = blockIdx.x * 256 + threadIdx.x;   // 0 .. 262143
    int i    = id & 15;
    int lane = (id >> 4) & 31;
    int ct   = (id >> 9) & 31;
    int ks   = (id >> 14) & 15;

    int K   = ks * 32 + ((lane >> 4) << 3) + (i & 7) + ((i >> 3) << 4);
    int col = ct * 16 + (lane & 15);

    int p = (K * col) & (NFFT - 1);              // phase index mod 512
    float ang = -6.28318530717958647692f * (float)p * (1.0f / (float)NFFT);
    float c = cosf(ang);                         // real coeff of e^{-i*2pi*K*col/N}
    float s = sinf(ang);                         // imag coeff

    _Float16 crh = (_Float16)c;
    _Float16 crl = (_Float16)(c - (float)crh);
    _Float16 cih = (_Float16)s;
    _Float16 cil = (_Float16)(s - (float)cih);

    size_t base = (size_t)(((ks * 32 + ct) * 32 + lane) * 16 + i);
    ws[0 * (size_t)BF_MAT + base] = crh;
    ws[1 * (size_t)BF_MAT + base] = crl;
    ws[2 * (size_t)BF_MAT + base] = cih;
    ws[3 * (size_t)BF_MAT + base] = cil;
}

// Load a 16x32 f16 A-fragment from an LDS row (ISA layout: lane kbase 0/8,
// halves 0..7 -> K kbase+0..7, halves 8..15 -> K kbase+16..23).
__device__ inline v16h load_a_frag(const _Float16* __restrict__ rowp, int ka) {
    v8h a0 = *(const v8h*)(rowp + ka);
    v8h a1 = *(const v8h*)(rowp + ka + 16);
    return __builtin_shufflevector(a0, a1, 0, 1, 2, 3, 4, 5, 6, 7,
                                           8, 9, 10, 11, 12, 13, 14, 15);
}

__device__ inline v8f wmma_f16(v16h a, v16h b, v8f c) {
    return __builtin_amdgcn_wmma_f32_16x16x32_f16(false, a, false, b,
                                                  (short)0, c, false, false);
}

// ---------------------------------------------------------------------------
// Main GEMM: block = 256 threads (8 wave32), 64 frames per block.
// LDS: A_hi[64][520], A_lo[64][520] halves (130 KB; <=2 blocks per 320KB WGP).
// Wave wv: row-tile pair rtg = wv>>2 (rows rtg*32 .. rtg*32+31),
//          col tiles ct = (wv&3), (wv&3)+4, ..., 28  (8 iterations).
// Per K-step: 4 B-frag loads (global b128, L2-hot) + 2x2 A-frag LDS loads,
// 12 v_wmma_f32_16x16x32_f16 (3 split terms x {rx,ix} x 2 row tiles).
// Live VGPRs ~= 32 acc + 32 B + 32 A + addr  -> no spills.
// ---------------------------------------------------------------------------
__global__ void stft_wmma(const float* __restrict__ x,
                          const float* __restrict__ win,
                          const _Float16* __restrict__ bfr,
                          float* __restrict__ out) {
    extern __shared__ _Float16 smem[];           // [2][RPB][LDSS]
    _Float16* Ah = smem;
    _Float16* Al = smem + (size_t)RPB * LDSS;

    const int tid      = threadIdx.x;
    const int rowBlock = blockIdx.x * RPB;

    // ---- Stage windowed frames into LDS as hi/lo f16 (coalesced on x) ----
    for (int idx = tid; idx < RPB * NFFT; idx += 256) {
        int row = idx >> 9;          // 0..63
        int kk  = idx & (NFFT - 1);  // 0..511
        int r   = rowBlock + row;    // global frame index
        int b   = r >> 11;           // batch
        int t   = r & (NUMF - 1);    // frame-in-batch
        int src = t * HLEN + kk;
        float v = 0.0f;
        if (src < SLEN) v = x[(size_t)b * SLEN + src] * win[kk];
        _Float16 h = (_Float16)v;
        Ah[row * LDSS + kk] = h;
        Al[row * LDSS + kk] = (_Float16)(v - (float)h);
    }
    __syncthreads();

    const int lane = tid & 31;
    const int wv   = tid >> 5;                    // wave id 0..7
    const int rtg  = wv >> 2;                     // row-tile group: 0 or 1
    const int mrow = lane & 15;
    const int kb0  = (lane >> 4) << 3;            // A-fragment K base: 0 or 8
    const int rsel = (lane >> 4) << 3;            // D layout: +8 rows, hi lanes

    // LDS row pointers for this wave's two row tiles (rows rtg*32+mrow, +16).
    const _Float16* __restrict__ rh0 = Ah + (size_t)(rtg * 32 + mrow) * LDSS;
    const _Float16* __restrict__ rl0 = Al + (size_t)(rtg * 32 + mrow) * LDSS;
    const _Float16* __restrict__ rh1 = rh0 + (size_t)16 * LDSS;
    const _Float16* __restrict__ rl1 = rl0 + (size_t)16 * LDSS;

    float* __restrict__ out_r = out;
    float* __restrict__ out_i = out + (size_t)ROWS * NFFT;

    const v8f zero = {0.f, 0.f, 0.f, 0.f, 0.f, 0.f, 0.f, 0.f};

    for (int ct = (wv & 3); ct < 32; ct += 4) {
        v8f aR0 = zero, aR1 = zero, aI0 = zero, aI1 = zero;

        const size_t bbase = (size_t)((ct * 32 + lane) * 16);

        for (int ks = 0; ks < 16; ++ks) {
            const size_t bo = (size_t)ks * 16384 + bbase;
            v16h bRh = *(const v16h*)(bfr + 0 * (size_t)BF_MAT + bo);
            v16h bRl = *(const v16h*)(bfr + 1 * (size_t)BF_MAT + bo);
            v16h bIh = *(const v16h*)(bfr + 2 * (size_t)BF_MAT + bo);
            v16h bIl = *(const v16h*)(bfr + 3 * (size_t)BF_MAT + bo);

            const int ka = ks * 32 + kb0;

            // ---- row tile 0 ----
            {
                v16h ah = load_a_frag(rh0, ka);
                v16h al = load_a_frag(rl0, ka);
                aR0 = wmma_f16(ah, bRh, aR0);
                aR0 = wmma_f16(ah, bRl, aR0);
                aR0 = wmma_f16(al, bRh, aR0);
                aI0 = wmma_f16(ah, bIh, aI0);
                aI0 = wmma_f16(ah, bIl, aI0);
                aI0 = wmma_f16(al, bIh, aI0);
            }
            // ---- row tile 1 ----
            {
                v16h ah = load_a_frag(rh1, ka);
                v16h al = load_a_frag(rl1, ka);
                aR1 = wmma_f16(ah, bRh, aR1);
                aR1 = wmma_f16(ah, bRl, aR1);
                aR1 = wmma_f16(al, bRh, aR1);
                aI1 = wmma_f16(ah, bIh, aI1);
                aI1 = wmma_f16(ah, bIl, aI1);
                aI1 = wmma_f16(al, bIh, aI1);
            }
        }

        // ---- Store D (ISA f32 C/D layout: VGPR j -> row j / j+8) ----
        const int col   = ct * 16 + mrow;
        const int rowt0 = rowBlock + rtg * 32 + rsel;         // tile 0 base
        const int rowt1 = rowt0 + 16;                         // tile 1 base
#pragma unroll
        for (int j = 0; j < 8; ++j) {
            const size_t o0 = (size_t)(rowt0 + j) * NFFT + col;
            const size_t o1 = (size_t)(rowt1 + j) * NFFT + col;
            out_r[o0] = aR0[j];
            out_i[o0] = aI0[j];
            out_r[o1] = aR1[j];
            out_i[o1] = aI1[j];
        }
    }
}

extern "C" void kernel_launch(void* const* d_in, const int* in_sizes, int n_in,
                              void* d_out, int out_size, void* d_ws, size_t ws_size,
                              hipStream_t stream) {
    const float* x   = (const float*)d_in[0];   // (16, 262144) f32
    const float* win = (const float*)d_in[1];   // (512,) f32
    _Float16* bfr = (_Float16*)d_ws;            // 2 MB of B-fragments

    // 1) Build DFT B-fragments (deterministic, rebuilt every call).
    build_bfrags<<<1024, 256, 0, stream>>>(bfr);

    // 2) Framed split-f16 WMMA GEMM producing rx then ix.
    const size_t smem_bytes = 2ull * RPB * LDSS * sizeof(_Float16); // 133,120 B
    stft_wmma<<<ROWS / RPB, 256, smem_bytes, stream>>>(x, win, bfr, (float*)d_out);
}
